// BMMS8TS8NF32T_61297773248991
// MI455X (gfx1250) — 8-bit matrix operand order repaired and checked by reading, compiled, not run on hardware
//
#include <hip/hip_runtime.h>
#include <stdint.h>

typedef int v8i __attribute__((ext_vector_type(8)));

#define TILE_M 64
#define TILE_N 128

__global__ __launch_bounds__(256)
void bmm_i8_wmma_kernel(const int8_t* __restrict__ X,
                        const int8_t* __restrict__ Y,
                        const float* __restrict__ pAzp,
                        const float* __restrict__ pBzp,
                        const float* __restrict__ pAlpha,
                        float* __restrict__ Out)
{
    const int tid = threadIdx.x;
    const int b   = blockIdx.z;
    const int m0  = blockIdx.y * TILE_M;
    const int n0  = blockIdx.x * TILE_N;

    __shared__ __align__(16) uint8_t s_a [TILE_M * 64];
    __shared__ __align__(16) uint8_t s_bT[TILE_N * 64];

    {
        const int8_t* src = X + ((size_t)b << 16) + (size_t)m0 * 64;
        *(int4*)(s_a + tid * 16) = *(const int4*)(src + tid * 16);
    }

    {
        const int8_t* srcb = Y + ((size_t)b << 16) + n0;
        const int cg  = tid & 31;
        const int kg0 = tid >> 5;
        #pragma unroll
        for (int it = 0; it < 2; ++it) {
            const int kg = kg0 + it * 8;
            uint32_t r0 = *(const uint32_t*)(srcb + (size_t)(kg * 4 + 0) * 1024 + cg * 4);
            uint32_t r1 = *(const uint32_t*)(srcb + (size_t)(kg * 4 + 1) * 1024 + cg * 4);
            uint32_t r2 = *(const uint32_t*)(srcb + (size_t)(kg * 4 + 2) * 1024 + cg * 4);
            uint32_t r3 = *(const uint32_t*)(srcb + (size_t)(kg * 4 + 3) * 1024 + cg * 4);
            uint32_t p01lo = __builtin_amdgcn_perm(r1, r0, 0x05010400u);
            uint32_t p01hi = __builtin_amdgcn_perm(r1, r0, 0x07030602u);
            uint32_t p23lo = __builtin_amdgcn_perm(r3, r2, 0x05010400u);
            uint32_t p23hi = __builtin_amdgcn_perm(r3, r2, 0x07030602u);
            uint32_t o0 = __builtin_amdgcn_perm(p23lo, p01lo, 0x05040100u);
            uint32_t o1 = __builtin_amdgcn_perm(p23lo, p01lo, 0x07060302u);
            uint32_t o2 = __builtin_amdgcn_perm(p23hi, p01hi, 0x05040100u);
            uint32_t o3 = __builtin_amdgcn_perm(p23hi, p01hi, 0x07060302u);
            *(uint32_t*)(s_bT + (cg * 4 + 0) * 64 + kg * 4) = o0;
            *(uint32_t*)(s_bT + (cg * 4 + 1) * 64 + kg * 4) = o1;
            *(uint32_t*)(s_bT + (cg * 4 + 2) * 64 + kg * 4) = o2;
            *(uint32_t*)(s_bT + (cg * 4 + 3) * 64 + kg * 4) = o3;
        }
    }

    __syncthreads();

    const int wave  = tid >> 5;
    const int lane  = tid & 31;
    const int wm    = wave >> 2;
    const int wn    = wave & 3;
    const int lm    = lane & 15;
    const int halfk = lane >> 4;

    v8i afr[2];
    #pragma unroll
    for (int f = 0; f < 2; ++f) {
        const int row = wm * 32 + f * 16 + lm;
        const uint8_t* base = s_a + row * 64 + halfk * 16;
        #pragma unroll
        for (int j = 0; j < 4; ++j) {
            uint2 d = *(const uint2*)(base + (j & 1) * 8 + (j >> 1) * 32);
            afr[f][2 * j]     = (int)d.x;
            afr[f][2 * j + 1] = (int)d.y;
        }
    }

    v8i bfr[2];
    #pragma unroll
    for (int g = 0; g < 2; ++g) {
        const int col = wn * 32 + g * 16 + lm;
        const uint8_t* base = s_bT + col * 64 + halfk * 16;
        int4 lo = *(const int4*)(base);
        int4 hi = *(const int4*)(base + 32);
        bfr[g][0] = lo.x; bfr[g][1] = lo.y; bfr[g][2] = lo.z; bfr[g][3] = lo.w;
        bfr[g][4] = hi.x; bfr[g][5] = hi.y; bfr[g][6] = hi.z; bfr[g][7] = hi.w;
    }

    const v8i zero = {0, 0, 0, 0, 0, 0, 0, 0};
    const int oc = 0x01010101;
    const v8i ones = {oc, oc, oc, oc, oc, oc, oc, oc};

    v8i c1[2][2], csr[2], csc[2];
    #pragma unroll
    for (int f = 0; f < 2; ++f)
        csr[f] = __builtin_amdgcn_wmma_i32_16x16x64_iu8(true, afr[f], true, ones, zero, false, false);
    #pragma unroll
    for (int g = 0; g < 2; ++g)
        csc[g] = __builtin_amdgcn_wmma_i32_16x16x64_iu8(true, ones, true, bfr[g], zero, false, false);
    #pragma unroll
    for (int f = 0; f < 2; ++f)
        #pragma unroll
        for (int g = 0; g < 2; ++g)
            c1[f][g] = __builtin_amdgcn_wmma_i32_16x16x64_iu8(true, afr[f], true, bfr[g], zero, false, false);

    const float az = *pAzp;
    const float bz = *pBzp;
    const float al = *pAlpha;
    const float kc = 64.0f * az * bz;
    float* outB = Out + ((size_t)b << 20);

    #pragma unroll
    for (int f = 0; f < 2; ++f) {
        const int rbase = m0 + wm * 32 + f * 16 + 8 * halfk;
        #pragma unroll
        for (int g = 0; g < 2; ++g) {
            const int cidx = n0 + wn * 32 + g * 16 + lm;
            #pragma unroll
            for (int i = 0; i < 8; ++i) {
                float v = ((float)c1[f][g][i]
                           - az * (float)csc[g][i]
                           - bz * (float)csr[f][i]
                           + kc) * al;
                outB[(size_t)(rbase + i) * 1024 + cidx] = v;
            }
        }
    }
}

extern "C" void kernel_launch(void* const* d_in, const int* in_sizes, int n_in,
                              void* d_out, int out_size, void* d_ws, size_t ws_size,
                              hipStream_t stream) {
    (void)in_sizes; (void)n_in; (void)out_size; (void)d_ws; (void)ws_size;
    const int8_t* x     = (const int8_t*)d_in[0];
    const int8_t* y     = (const int8_t*)d_in[1];
    const float*  a_zp  = (const float*)d_in[2];
    const float*  b_zp  = (const float*)d_in[3];
    const float*  alpha = (const float*)d_in[4];
    float* out = (float*)d_out;

    dim3 grid(1024 / TILE_N, 1024 / TILE_M, 96);
    bmm_i8_wmma_kernel<<<grid, 256, 0, stream>>>(x, y, a_zp, b_zp, alpha, out);
}
